// SignNetLayer_20555713479321
// MI455X (gfx1250) — matrix operand order repaired and checked by reading, not compiled, not run on hardware
//
#include <hip/hip_runtime.h>
#include <hip/hip_bf16.h>

#define N_NODES 100000
#define N_EDGES 1600000
#define DIM_ED  16
#define DIM_LH  96
#define DIM_G   384            // 4*LH
#define NTILES  (N_NODES/16)   // 6250
#define ETILES  (N_EDGES/16)   // 100000

typedef __attribute__((ext_vector_type(16))) _Float16 v16h;
typedef __attribute__((ext_vector_type(8)))  _Float16 v8h;
typedef __attribute__((ext_vector_type(8)))  float    v8f;

static __device__ __forceinline__ v8f wmma32f16(v16h a, v16h b, v8f c) {
  return __builtin_amdgcn_wmma_f32_16x16x32_f16(false, a, false, b, (short)0, c,
                                                false, false);
}

// B fragment from pre-transposed weight BT = W^T, row-major [Ncols x ldk] f16.
// Lane holds B[kb + (lane/16)*8 + h][col0 + lane%16] for h = 0..7, B[kb + 16 + (lane/16)*8 + h-8][..] for h = 8..15.
static __device__ __forceinline__ v16h ld_bfrag(const _Float16* BT, int ldk,
                                                int col0, int kb) {
  int lane = threadIdx.x & 31;
  const _Float16* p = BT + (size_t)(col0 + (lane & 15)) * ldk + kb + ((lane >> 4) << 3);
  v8h lo = *(const v8h*)p;
  v8h hi = *(const v8h*)(p + 16);
  v16h r;
#pragma unroll
  for (int i = 0; i < 8; ++i) { r[i] = lo[i]; r[i + 8] = hi[i]; }
  return r;
}

// A fragment from row-major f16 [rows x lda]: lane = row, halves 0..7 -> K=kb+kh..,
// halves 8..15 -> K=kb+16+kh.., kh = (lane/16)*8.
static __device__ __forceinline__ v16h ld_afrag16(const _Float16* A, int lda,
                                                  int row0, int kb) {
  int lane = threadIdx.x & 31;
  const _Float16* p = A + (size_t)(row0 + (lane & 15)) * lda + kb + ((lane >> 4) << 3);
  v8h lo = *(const v8h*)p;
  v8h hi = *(const v8h*)(p + 16);
  v16h r;
#pragma unroll
  for (int i = 0; i < 8; ++i) { r[i] = lo[i]; r[i + 8] = hi[i]; }
  return r;
}

// ---------------- utility kernels ----------------
__global__ void zero_f32(float* p, size_t n) {
  size_t i = (size_t)blockIdx.x * blockDim.x + threadIdx.x;
  size_t st = (size_t)gridDim.x * blockDim.x;
  for (; i < n; i += st) p[i] = 0.f;
}

// BT[n][k] = (k<K) ? f16(W[k*Ncols+n]) : 0   (weights -> f16, transposed, K padded)
__global__ void wprep(const float* __restrict__ W, _Float16* __restrict__ BT,
                      int K, int Ncols, int Kpad) {
  int idx = blockIdx.x * blockDim.x + threadIdx.x;
  if (idx >= Ncols * Kpad) return;
  int n = idx / Kpad, k = idx % Kpad;
  BT[idx] = (k < K) ? (_Float16)W[(size_t)k * Ncols + n] : (_Float16)0.f;
}

// ---------------- GINE edge phase: e = eattr@We+be; scatter relu(x[src]+e) ----
__global__ void edge_kernel(const float* __restrict__ eattr,
                            const long long* __restrict__ ei,
                            const float* __restrict__ xsrc, float sign, int ic,
                            const _Float16* __restrict__ WeT,
                            const float* __restrict__ be,
                            float* __restrict__ agg) {
  int wave = (int)((blockIdx.x * blockDim.x + threadIdx.x) >> 5);
  int lane = threadIdx.x & 31;
  if (wave >= ETILES) return;
  int e0 = wave * 16;
  // A fragment: 16 edges x 16 features (K padded to 32 with zeros)
  int arow = e0 + (lane & 15);
  int kh = (lane >> 4) * 8;
  const float4* pa = (const float4*)(eattr + (size_t)arow * DIM_ED + kh);
  float4 f0 = pa[0], f1 = pa[1];
  v16h a;
  a[0]=(_Float16)f0.x; a[1]=(_Float16)f0.y; a[2]=(_Float16)f0.z; a[3]=(_Float16)f0.w;
  a[4]=(_Float16)f1.x; a[5]=(_Float16)f1.y; a[6]=(_Float16)f1.z; a[7]=(_Float16)f1.w;
#pragma unroll
  for (int i = 8; i < 16; ++i) a[i] = (_Float16)0.f;
  int rb = (lane >> 4) * 8;
  long long s8[8], d8[8];
#pragma unroll
  for (int v = 0; v < 8; ++v) {
    s8[v] = ei[e0 + rb + v];
    d8[v] = ei[(size_t)N_EDGES + e0 + rb + v];
  }
  int nnt = ic >> 4;
  for (int nt = 0; nt < nnt; ++nt) {
    v8f c = {};
    c = wmma32f16(a, ld_bfrag(WeT, 32, nt * 16, 0), c);
    int col = nt * 16 + (lane & 15);
    float bev = be[col];
#pragma unroll
    for (int v = 0; v < 8; ++v) {
      float ev = c[v] + bev;
      float xv = sign * xsrc[(size_t)s8[v] * ic + col];
      float m = fmaxf(xv + ev, 0.f);
      atomicAdd(agg + (size_t)d8[v] * ic + col, m);  // segment_sum
    }
  }
}

// ---------------- GINE node MLP: xs[l] = relu(relu((x+agg)@W1+b1)@W2+b2) -----
__global__ void node_kernel(const float* __restrict__ xin, float sign, int ic,
                            const float* __restrict__ agg,
                            const _Float16* __restrict__ W1T, int kpad1,
                            const float* __restrict__ b1,
                            const _Float16* __restrict__ W2T,
                            const float* __restrict__ b2,
                            float* __restrict__ xout,
                            _Float16* __restrict__ xouth) {
  __shared__ _Float16 st[2][16 * 64];
  int wid = threadIdx.x >> 5, lane = threadIdx.x & 31;
  int tile = blockIdx.x * 2 + wid;
  int n0 = tile * 16;
  int arow = n0 + (lane & 15);
  int kh = (lane >> 4) * 8;
  size_t roff = (size_t)arow * ic;
  v16h a0 = {}, a1 = {};
#pragma unroll
  for (int i = 0; i < 8; ++i) {
    int k0 = kh + i, k1 = 16 + kh + i;
    a0[i]     = (k0 < ic) ? (_Float16)(sign * xin[roff + k0] + agg[roff + k0]) : (_Float16)0.f;
    a0[i + 8] = (k1 < ic) ? (_Float16)(sign * xin[roff + k1] + agg[roff + k1]) : (_Float16)0.f;
  }
  if (kpad1 == 64) {
#pragma unroll
    for (int i = 0; i < 8; ++i) {
      int k0 = 32 + kh + i, k1 = 48 + kh + i;
      a1[i]     = (_Float16)(sign * xin[roff + k0] + agg[roff + k0]);
      a1[i + 8] = (_Float16)(sign * xin[roff + k1] + agg[roff + k1]);
    }
  }
  int rb = (lane >> 4) * 8;
  for (int nt = 0; nt < 4; ++nt) {
    v8f c = {};
    c = wmma32f16(a0, ld_bfrag(W1T, kpad1, nt * 16, 0), c);
    if (kpad1 == 64) c = wmma32f16(a1, ld_bfrag(W1T, kpad1, nt * 16, 32), c);
    int col = nt * 16 + (lane & 15);
    float bb = b1[col];
#pragma unroll
    for (int v = 0; v < 8; ++v)
      st[wid][(rb + v) * 64 + col] = (_Float16)fmaxf(c[v] + bb, 0.f);
  }
  __syncthreads();                                  // C-layout -> A-layout via LDS
  v16h t0 = ld_afrag16(&st[wid][0], 64, 0, 0);
  v16h t1 = ld_afrag16(&st[wid][0], 64, 0, 32);
  for (int nt = 0; nt < 4; ++nt) {
    v8f c = {};
    c = wmma32f16(t0, ld_bfrag(W2T, 64, nt * 16, 0), c);
    c = wmma32f16(t1, ld_bfrag(W2T, 64, nt * 16, 32), c);
    int col = nt * 16 + (lane & 15);
    float bb = b2[col];
#pragma unroll
    for (int v = 0; v < 8; ++v) {
      float y = fmaxf(c[v] + bb, 0.f);
      size_t o = (size_t)(n0 + rb + v) * 64 + col;
      xout[o] = y;
      xouth[o] = (_Float16)y;
    }
  }
}

// ---------------- one bi-LSTM timestep (fwd uses t, bwd uses 2-t) ------------
__global__ void lstm_step(const _Float16* __restrict__ xtf,
                          const _Float16* __restrict__ xtb,
                          const _Float16* __restrict__ WihTf, const _Float16* __restrict__ WhhTf,
                          const float* __restrict__ bif, const float* __restrict__ bhf,
                          const _Float16* __restrict__ WihTb, const _Float16* __restrict__ WhhTb,
                          const float* __restrict__ bib, const float* __restrict__ bhb,
                          _Float16* __restrict__ hf16, float* __restrict__ cf,
                          _Float16* __restrict__ hb16, float* __restrict__ cb,
                          const float* __restrict__ watt, const float* __restrict__ batt,
                          float* __restrict__ logits, int t) {
  __shared__ float g[2][16 * DIM_G];
  __shared__ float nl[2][16];
  int wid = threadIdx.x >> 5, lane = threadIdx.x & 31;
  int tile = blockIdx.x * 2 + wid;
  int n0 = tile * 16;
  int rb = (lane >> 4) * 8;
  for (int dir = 0; dir < 2; ++dir) {
    const _Float16* xt = dir ? xtb : xtf;
    const _Float16* WihT = dir ? WihTb : WihTf;
    const _Float16* WhhT = dir ? WhhTb : WhhTf;
    const float* bi = dir ? bib : bif;
    const float* bh = dir ? bhb : bhf;
    _Float16* h16 = dir ? hb16 : hf16;
    float* cc = dir ? cb : cf;
    v16h ax0 = ld_afrag16(xt, 64, n0, 0);
    v16h ax1 = ld_afrag16(xt, 64, n0, 32);
    v16h ah0 = ld_afrag16(h16, 96, n0, 0);
    v16h ah1 = ld_afrag16(h16, 96, n0, 32);
    v16h ah2 = ld_afrag16(h16, 96, n0, 64);
    for (int nt = 0; nt < 24; ++nt) {        // gates g = x@Wih + h@Whh + b
      v8f c = {};
      c = wmma32f16(ax0, ld_bfrag(WihT, 64, nt * 16, 0), c);
      c = wmma32f16(ax1, ld_bfrag(WihT, 64, nt * 16, 32), c);
      c = wmma32f16(ah0, ld_bfrag(WhhT, 96, nt * 16, 0), c);
      c = wmma32f16(ah1, ld_bfrag(WhhT, 96, nt * 16, 32), c);
      c = wmma32f16(ah2, ld_bfrag(WhhT, 96, nt * 16, 64), c);
      int col = nt * 16 + (lane & 15);
      float bb = bi[col] + bh[col];
#pragma unroll
      for (int v = 0; v < 8; ++v)
        g[wid][(rb + v) * DIM_G + col] = c[v] + bb;
    }
    if (lane < 16) nl[wid][lane] = 0.f;
    __syncthreads();
    for (int idx = lane; idx < 16 * DIM_LH; idx += 32) {
      int r = idx / DIM_LH, ch = idx % DIM_LH;
      const float* gr = &g[wid][r * DIM_G];
      float gi = gr[ch], gf = gr[DIM_LH + ch], gg = gr[2 * DIM_LH + ch], go = gr[3 * DIM_LH + ch];
      size_t off = (size_t)(n0 + r) * DIM_LH + ch;
      float cv = cc[off];
      float si = 1.f / (1.f + __expf(-gi));
      float sf = 1.f / (1.f + __expf(-gf));
      float so = 1.f / (1.f + __expf(-go));
      float cn = sf * cv + si * tanhf(gg);
      float hv = so * tanhf(cn);
      cc[off] = cn;
      h16[off] = (_Float16)hv;
      atomicAdd(&nl[wid][r], hv * watt[dir * DIM_LH + ch]);  // h . Watt (per node)
    }
    __syncthreads();
    if (lane < 16) {
      int slot = dir ? (2 - t) : t;
      logits[(size_t)slot * N_NODES + n0 + lane] += nl[wid][lane] + (dir ? 0.f : batt[0]);
    }
    __syncthreads();
  }
}

// ---------------- attention softmax + JK weighted sum + Wl -------------------
__global__ void attn_kernel(const float* __restrict__ logits,
                            const float* __restrict__ x0, const float* __restrict__ x1,
                            const float* __restrict__ x2,
                            const _Float16* __restrict__ WlT,
                            const float* __restrict__ bl,
                            float* __restrict__ phis) {
  int wid = threadIdx.x >> 5, lane = threadIdx.x & 31;
  int tile = blockIdx.x * 2 + wid;
  int n0 = tile * 16;
  int n = n0 + (lane & 15);
  float l0 = logits[n], l1 = logits[(size_t)N_NODES + n], l2 = logits[2 * (size_t)N_NODES + n];
  float m = fmaxf(l0, fmaxf(l1, l2));
  float e0 = __expf(l0 - m), e1 = __expf(l1 - m), e2 = __expf(l2 - m);
  float inv = 1.f / (e0 + e1 + e2);
  float a0w = e0 * inv, a1w = e1 * inv, a2w = e2 * inv;
  int kh = (lane >> 4) * 8;
  size_t roff = (size_t)n * 64;
  v16h A0 = {}, A1 = {};
#pragma unroll
  for (int i = 0; i < 8; ++i) {
    int k0 = kh + i, k1 = 16 + kh + i, k2 = 32 + kh + i, k3 = 48 + kh + i;
    A0[i]     = (_Float16)(a0w * x0[roff + k0] + a1w * x1[roff + k0] + a2w * x2[roff + k0]);
    A0[i + 8] = (_Float16)(a0w * x0[roff + k1] + a1w * x1[roff + k1] + a2w * x2[roff + k1]);
    A1[i]     = (_Float16)(a0w * x0[roff + k2] + a1w * x1[roff + k2] + a2w * x2[roff + k2]);
    A1[i + 8] = (_Float16)(a0w * x0[roff + k3] + a1w * x1[roff + k3] + a2w * x2[roff + k3]);
  }
  int rb = (lane >> 4) * 8;
  for (int nt = 0; nt < 4; ++nt) {
    v8f c = {};
    c = wmma32f16(A0, ld_bfrag(WlT, 64, nt * 16, 0), c);
    c = wmma32f16(A1, ld_bfrag(WlT, 64, nt * 16, 32), c);
    int col = nt * 16 + (lane & 15);
    float bb = bl[col];
#pragma unroll
    for (int v = 0; v < 8; ++v)
      phis[(size_t)(n0 + rb + v) * 64 + col] += c[v] + bb;   // sum over +/- passes
  }
}

// ---------------- rho stage 1: h1 = phi@Wr1+br1, + BN stats ------------------
__global__ void rho1_kernel(const float* __restrict__ phis,
                            const _Float16* __restrict__ Wr1T,
                            const float* __restrict__ br1,
                            float* __restrict__ h1, float* __restrict__ stats) {
  int wid = threadIdx.x >> 5, lane = threadIdx.x & 31;
  int tile = blockIdx.x * 2 + wid;
  int n0 = tile * 16;
  int kh = (lane >> 4) * 8;
  size_t roff = (size_t)(n0 + (lane & 15)) * 64;
  v16h A0 = {}, A1 = {};
#pragma unroll
  for (int i = 0; i < 8; ++i) {
    A0[i]     = (_Float16)phis[roff + kh + i];
    A0[i + 8] = (_Float16)phis[roff + 16 + kh + i];
    A1[i]     = (_Float16)phis[roff + 32 + kh + i];
    A1[i + 8] = (_Float16)phis[roff + 48 + kh + i];
  }
  int rb = (lane >> 4) * 8;
  for (int nt = 0; nt < 4; ++nt) {
    v8f c = {};
    c = wmma32f16(A0, ld_bfrag(Wr1T, 64, nt * 16, 0), c);
    c = wmma32f16(A1, ld_bfrag(Wr1T, 64, nt * 16, 32), c);
    int col = nt * 16 + (lane & 15);
    float bb = br1[col];
    float ps = 0.f, pq = 0.f;
#pragma unroll
    for (int v = 0; v < 8; ++v) {
      float y = c[v] + bb;
      h1[(size_t)(n0 + rb + v) * 64 + col] = y;
      ps += y; pq += y * y;
    }
    atomicAdd(&stats[col], ps);
    atomicAdd(&stats[64 + col], pq);
  }
}

__global__ void bn_final(const float* __restrict__ stats,
                         const float* __restrict__ gamma,
                         const float* __restrict__ beta, float* __restrict__ bnp) {
  int c = threadIdx.x;
  if (c < 64) {
    float mean = stats[c] / (float)N_NODES;
    float var = stats[64 + c] / (float)N_NODES - mean * mean;
    float sc = gamma[c] * rsqrtf(var + 1e-5f);
    bnp[c] = sc;
    bnp[64 + c] = beta[c] - mean * sc;
  }
}

// ---------------- rho stage 2: out = relu(BN(h1))@Wr2+br2 --------------------
__global__ void rho2_kernel(const float* __restrict__ h1,
                            const float* __restrict__ bnp,
                            const _Float16* __restrict__ Wr2T,
                            const float* __restrict__ br2,
                            float* __restrict__ out) {
  int wid = threadIdx.x >> 5, lane = threadIdx.x & 31;
  int tile = blockIdx.x * 2 + wid;
  int n0 = tile * 16;
  int kh = (lane >> 4) * 8;
  size_t roff = (size_t)(n0 + (lane & 15)) * 64;
  v16h A0 = {}, A1 = {};
#pragma unroll
  for (int i = 0; i < 8; ++i) {
    int k0 = kh + i, k1 = 16 + kh + i, k2 = 32 + kh + i, k3 = 48 + kh + i;
    A0[i]     = (_Float16)fmaxf(h1[roff + k0] * bnp[k0] + bnp[64 + k0], 0.f);
    A0[i + 8] = (_Float16)fmaxf(h1[roff + k1] * bnp[k1] + bnp[64 + k1], 0.f);
    A1[i]     = (_Float16)fmaxf(h1[roff + k2] * bnp[k2] + bnp[64 + k2], 0.f);
    A1[i + 8] = (_Float16)fmaxf(h1[roff + k3] * bnp[k3] + bnp[64 + k3], 0.f);
  }
  int rb = (lane >> 4) * 8;
  for (int nt = 0; nt < 4; ++nt) {
    v8f c = {};
    c = wmma32f16(A0, ld_bfrag(Wr2T, 64, nt * 16, 0), c);
    c = wmma32f16(A1, ld_bfrag(Wr2T, 64, nt * 16, 32), c);
    int col = nt * 16 + (lane & 15);
    float bb = br2[col];
#pragma unroll
    for (int v = 0; v < 8; ++v)
      out[(size_t)(n0 + rb + v) * 64 + col] = c[v] + bb;
  }
}

// ---------------- host launch helpers ----------------------------------------
static void zero_launch(float* p, size_t n, hipStream_t s) {
  int blocks = (int)((n + 255) / 256);
  if (blocks > 4096) blocks = 4096;
  zero_f32<<<blocks, 256, 0, s>>>(p, n);
}
static void prep_launch(const float* W, _Float16* BT, int K, int Nc, int Kpad,
                        hipStream_t s) {
  int total = Nc * Kpad;
  wprep<<<(total + 255) / 256, 256, 0, s>>>(W, BT, K, Nc, Kpad);
}

extern "C" void kernel_launch(void* const* d_in, const int* in_sizes, int n_in,
                              void* d_out, int out_size, void* d_ws, size_t ws_size,
                              hipStream_t stream) {
  (void)in_sizes; (void)n_in; (void)out_size; (void)ws_size;
  // dict-flatten (sorted keys) order of setup_inputs()
  const float*     pos   = (const float*)d_in[0];
  const long long* ei    = (const long long*)d_in[1];
  const float*     eattr = (const float*)d_in[2];
  const float* Wl  = (const float*)d_in[3];
  const float* Wr1 = (const float*)d_in[4];
  const float* Wr2 = (const float*)d_in[5];
  const float* beta = (const float*)d_in[6];
  const float* bl  = (const float*)d_in[7];
  const float* br1 = (const float*)d_in[8];
  const float* br2 = (const float*)d_in[9];
  const float* cW1[3] = {(const float*)d_in[10], (const float*)d_in[16], (const float*)d_in[22]};
  const float* cW2[3] = {(const float*)d_in[11], (const float*)d_in[17], (const float*)d_in[23]};
  const float* cWe[3] = {(const float*)d_in[12], (const float*)d_in[18], (const float*)d_in[24]};
  const float* cb1[3] = {(const float*)d_in[13], (const float*)d_in[19], (const float*)d_in[25]};
  const float* cb2[3] = {(const float*)d_in[14], (const float*)d_in[20], (const float*)d_in[26]};
  const float* cbe[3] = {(const float*)d_in[15], (const float*)d_in[21], (const float*)d_in[27]};
  const float* gamma = (const float*)d_in[28];
  const float* Watt  = (const float*)d_in[29];
  const float* Whh_b = (const float*)d_in[30];
  const float* Whh_f = (const float*)d_in[31];
  const float* Wih_b = (const float*)d_in[32];
  const float* Wih_f = (const float*)d_in[33];
  const float* batt  = (const float*)d_in[34];
  const float* bhh_b = (const float*)d_in[35];
  const float* bhh_f = (const float*)d_in[36];
  const float* bih_b = (const float*)d_in[37];
  const float* bih_f = (const float*)d_in[38];

  char* w = (char*)d_ws;
  auto take = [&](size_t bytes) -> char* {
    char* p = w; w += (bytes + 255) & ~(size_t)255; return p;
  };
  _Float16* WeT[3]; _Float16* W1T[3]; _Float16* W2T[3];
  WeT[0] = (_Float16*)take(16 * 32 * 2);
  WeT[1] = (_Float16*)take(64 * 32 * 2);
  WeT[2] = (_Float16*)take(64 * 32 * 2);
  W1T[0] = (_Float16*)take(64 * 32 * 2);
  W1T[1] = (_Float16*)take(64 * 64 * 2);
  W1T[2] = (_Float16*)take(64 * 64 * 2);
  for (int l = 0; l < 3; ++l) W2T[l] = (_Float16*)take(64 * 64 * 2);
  _Float16* WihT[2]; _Float16* WhhT[2];
  WihT[0] = (_Float16*)take(384 * 64 * 2);
  WihT[1] = (_Float16*)take(384 * 64 * 2);
  WhhT[0] = (_Float16*)take(384 * 96 * 2);
  WhhT[1] = (_Float16*)take(384 * 96 * 2);
  _Float16* WlT  = (_Float16*)take(64 * 64 * 2);
  _Float16* Wr1T = (_Float16*)take(64 * 64 * 2);
  _Float16* Wr2T = (_Float16*)take(64 * 64 * 2);
  float* agg = (float*)take((size_t)N_NODES * 64 * 4);
  float* xs[3]; _Float16* xsh[3];
  for (int l = 0; l < 3; ++l) xs[l]  = (float*)take((size_t)N_NODES * 64 * 4);
  for (int l = 0; l < 3; ++l) xsh[l] = (_Float16*)take((size_t)N_NODES * 64 * 2);
  _Float16* hh[2]; float* cst[2];
  hh[0]  = (_Float16*)take((size_t)N_NODES * 96 * 2);
  hh[1]  = (_Float16*)take((size_t)N_NODES * 96 * 2);
  cst[0] = (float*)take((size_t)N_NODES * 96 * 4);
  cst[1] = (float*)take((size_t)N_NODES * 96 * 4);
  float* logits = (float*)take(3 * (size_t)N_NODES * 4);
  float* phis   = (float*)take((size_t)N_NODES * 64 * 4);
  float* h1     = (float*)take((size_t)N_NODES * 64 * 4);
  float* stats  = (float*)take(128 * 4);
  float* bnp    = (float*)take(128 * 4);

  // weight prep (f16, transposed, K padded to WMMA multiples)
  prep_launch(cWe[0], WeT[0], 16, 16, 32, stream);
  prep_launch(cWe[1], WeT[1], 16, 64, 32, stream);
  prep_launch(cWe[2], WeT[2], 16, 64, 32, stream);
  prep_launch(cW1[0], W1T[0], 16, 64, 32, stream);
  prep_launch(cW1[1], W1T[1], 64, 64, 64, stream);
  prep_launch(cW1[2], W1T[2], 64, 64, 64, stream);
  for (int l = 0; l < 3; ++l) prep_launch(cW2[l], W2T[l], 64, 64, 64, stream);
  prep_launch(Wih_f, WihT[0], 64, 384, 64, stream);
  prep_launch(Wih_b, WihT[1], 64, 384, 64, stream);
  prep_launch(Whh_f, WhhT[0], 96, 384, 96, stream);
  prep_launch(Whh_b, WhhT[1], 96, 384, 96, stream);
  prep_launch(Wl,  WlT,  64, 64, 64, stream);
  prep_launch(Wr1, Wr1T, 64, 64, 64, stream);
  prep_launch(Wr2, Wr2T, 64, 64, 64, stream);

  zero_launch(phis, (size_t)N_NODES * 64, stream);

  for (int pass = 0; pass < 2; ++pass) {          // phi(+x) + phi(-x)
    float sgn = pass ? -1.f : 1.f;
    zero_launch(logits, 3 * (size_t)N_NODES, stream);
    zero_launch((float*)hh[0], (size_t)N_NODES * 48, stream);
    zero_launch((float*)hh[1], (size_t)N_NODES * 48, stream);
    zero_launch(cst[0], (size_t)N_NODES * 96, stream);
    zero_launch(cst[1], (size_t)N_NODES * 96, stream);
    const float* xcur = pos;
    for (int l = 0; l < 3; ++l) {
      int ic = l ? 64 : 16;
      float s = l ? 1.f : sgn;
      zero_launch(agg, (size_t)N_NODES * ic, stream);
      edge_kernel<<<ETILES / 4, 128, 0, stream>>>(eattr, ei, xcur, s, ic,
                                                  WeT[l], cbe[l], agg);
      node_kernel<<<NTILES / 2, 64, 0, stream>>>(xcur, s, ic, agg,
                                                 W1T[l], l ? 64 : 32, cb1[l],
                                                 W2T[l], cb2[l], xs[l], xsh[l]);
      xcur = xs[l];
    }
    for (int t = 0; t < 3; ++t)
      lstm_step<<<NTILES / 2, 64, 0, stream>>>(xsh[t], xsh[2 - t],
                                               WihT[0], WhhT[0], bih_f, bhh_f,
                                               WihT[1], WhhT[1], bih_b, bhh_b,
                                               hh[0], cst[0], hh[1], cst[1],
                                               Watt, batt, logits, t);
    attn_kernel<<<NTILES / 2, 64, 0, stream>>>(logits, xs[0], xs[1], xs[2],
                                               WlT, bl, phis);
  }
  zero_launch(stats, 128, stream);
  rho1_kernel<<<NTILES / 2, 64, 0, stream>>>(phis, Wr1T, br1, h1, stats);
  bn_final<<<1, 64, 0, stream>>>(stats, gamma, beta, bnp);
  rho2_kernel<<<NTILES / 2, 64, 0, stream>>>(h1, bnp, Wr2T, br2, (float*)d_out);
}
